// LSTMDecoder_80238579024342
// MI455X (gfx1250) — matrix operand order repaired and checked by reading, not compiled, not run on hardware
//
#include <hip/hip_runtime.h>
#include <hip/hip_bf16.h>

typedef __attribute__((ext_vector_type(16))) _Float16 v16h;
typedef __attribute__((ext_vector_type(8)))  _Float16 v8h;
typedef __attribute__((ext_vector_type(8)))  float    v8f;

#define LSTM_B   1024
#define LSTM_A   16
#define LSTM_T   128
#define LSTM_H   16
#define OUTW     (LSTM_A + LSTM_T)   // 144
#define WIN_P    17                  // padded window row pitch (bank-conflict free)

// gfx1250 has V_TANH_F32 (TRANS op). Use it when clang exposes the builtin;
// otherwise fall back to exp-based forms.
#if __has_builtin(__builtin_amdgcn_tanhf)
__device__ __forceinline__ float fast_tanh(float x) {
    return __builtin_amdgcn_tanhf(x);
}
__device__ __forceinline__ float fast_sigmoid(float x) {
    // sigmoid(x) = 0.5*tanh(x/2) + 0.5  (mul + tanh + fma)
    return __builtin_fmaf(0.5f, __builtin_amdgcn_tanhf(0.5f * x), 0.5f);
}
#else
__device__ __forceinline__ float fast_sigmoid(float x) {
    return 1.0f / (1.0f + __expf(-x));
}
__device__ __forceinline__ float fast_tanh(float x) {
    return 2.0f / (1.0f + __expf(-2.0f * x)) - 1.0f;
}
#endif

__global__ __launch_bounds__(32)
void lstm_decoder_wmma_kernel(const float* __restrict__ y,
                              const float* __restrict__ u,
                              const float* __restrict__ W_ih,
                              const float* __restrict__ W_hh,
                              const float* __restrict__ b_ih,
                              const float* __restrict__ b_hh,
                              const float* __restrict__ W_lin,
                              const float* __restrict__ b_lin,
                              const float* __restrict__ W_h0,
                              const float* __restrict__ b_h0,
                              const float* __restrict__ W_c0,
                              const float* __restrict__ b_c0,
                              float* __restrict__ out)
{
    // One wave32 per workgroup -> workgroup barriers degrade to S_NOP (+waitcnt).
    __shared__ float win_s[16 * WIN_P];                 // sliding window, circular cols
    __shared__ __align__(16) _Float16 h16A[256];        // h ping (16x16 f16, row-major)
    __shared__ __align__(16) _Float16 h16B[256];        // h pong

    const int lane = threadIdx.x & 31;
    const int tile = blockIdx.x;                        // 0..63
    const int bm   = tile * 16;                         // first batch row of tile
    const int n    = lane & 15;                         // N / column index
    const int hi   = lane >> 4;                         // lane half

    // ---- Per-lane scalar constants ----------------------------------------
    const float wlin = W_lin[n];
    const float blin = b_lin[0];
    const float bh0v = b_h0[n];
    const float bc0v = b_c0[n];

    // ---- W_hh as four augmented B fragments --------------------------------
    // B layout (16-bit, 32x16): element e<8 holds K=8*hi+e, e>=8 holds K=16+8*hi+(e-8).
    //   K=0..15  : W_hh^T        (elements 0..7 of both halves)
    //   K=16     : w_in[g*16+n]  (hi==0 lanes, element 8)
    //   K=17     : b_ih+b_hh     (hi==0 lanes, element 9)
    // Paired with A rows carrying [h[m][:], x[m], 1.0], the WMMA emits the
    // complete pre-activation gates.
    v16h Bg[4];
#pragma unroll
    for (int g = 0; g < 4; ++g) {
        const int j = g * 16 + n;
        v16h bf = {};
        {
            const float* row = W_hh + j * LSTM_H + 8 * hi;
#pragma unroll
            for (int e = 0; e < 8; ++e) bf[e] = (_Float16)row[e];
        }
        if (hi == 0) {
            bf[8] = (_Float16)W_ih[j];                 // W_ih is (64,1)
            bf[9] = (_Float16)(b_ih[j] + b_hh[j]);
        }
        Bg[g] = bf;
    }

    // ---- h0 / c0 = u @ W_{h,c}0^T + b via 8 chained WMMAs (K=128) ----------
    v8f hacc = {};
    v8f cacc = {};
#pragma unroll
    for (int q = 0; q < 4; ++q) {
        // A fragment from u (m = n = lane%16):
        // e<8: K = 32q + 8*hi + e ; e>=8: K = 32q + 16 + 8*hi + (e-8)
        const float* urow = u + (size_t)(bm + n) * 128;
        v16h a;
#pragma unroll
        for (int e = 0; e < 8; ++e) a[e]     = (_Float16)urow[32 * q + 8 * hi + e];
#pragma unroll
        for (int e = 0; e < 8; ++e) a[8 + e] = (_Float16)urow[32 * q + 16 + 8 * hi + e];

        v16h bh, bc;
#pragma unroll
        for (int e = 0; e < 16; ++e) {
            bh[e] = (_Float16)W_h0[(size_t)n * 128 + 32 * q + 8 * hi + (e & 7) + ((e >> 3) << 4)];
            bc[e] = (_Float16)W_c0[(size_t)n * 128 + 32 * q + 8 * hi + (e & 7) + ((e >> 3) << 4)];
        }
        hacc = __builtin_amdgcn_wmma_f32_16x16x32_f16(false, a, false, bh, (short)0, hacc, false, false);
        cacc = __builtin_amdgcn_wmma_f32_16x16x32_f16(false, a, false, bc, (short)0, cacc, false, false);
    }

    float cst[8], hcur[8];
#pragma unroll
    for (int r = 0; r < 8; ++r) {
        hcur[r] = hacc[r] + bh0v;
        cst[r]  = cacc[r] + bc0v;
    }

    // ---- Init window from y, copy y into out[:, 0:16] ----------------------
    for (int idx = lane; idx < 256; idx += 32) {
        const int m = idx >> 4, cc = idx & 15;
        const float v = y[(size_t)(bm + m) * LSTM_A + cc];
        win_s[m * WIN_P + cc] = v;
        out[(size_t)(bm + m) * OUTW + cc] = v;
    }
    // Initial h -> f16 ping buffer (D-layout rows m = r+8*hi, col n)
#pragma unroll
    for (int r = 0; r < 8; ++r)
        h16A[(r + 8 * hi) * 16 + n] = (_Float16)hcur[r];
    __syncthreads();

    // ---- Main recurrence ----------------------------------------------------
    int start = 0;   // circular window start column

    auto cell = [&](const _Float16* __restrict__ hr, _Float16* __restrict__ hw, int j) {
        const int col = (start + j) & 15;

        // A fragment: one b128 LDS load (row m = n, halves k0..k0+7, k0 = 8*hi),
        // augmented with x[m] (K=16) and 1.0 (K=17) in the hi==0 lanes.
        const v8h hv = *(const v8h*)(hr + n * 16 + 8 * hi);
        const float xv = win_s[n * WIN_P + col];        // broadcast pairwise
        v16h a = {};
#pragma unroll
        for (int e = 0; e < 8; ++e) a[e] = hv[e];
        a[8] = (hi == 0) ? (_Float16)xv   : (_Float16)0.0f;
        a[9] = (hi == 0) ? (_Float16)1.0f : (_Float16)0.0f;

        const v8f z = {};
        v8f gi = __builtin_amdgcn_wmma_f32_16x16x32_f16(false, a, false, Bg[0], (short)0, z, false, false);
        v8f gf = __builtin_amdgcn_wmma_f32_16x16x32_f16(false, a, false, Bg[1], (short)0, z, false, false);
        v8f gg = __builtin_amdgcn_wmma_f32_16x16x32_f16(false, a, false, Bg[2], (short)0, z, false, false);
        v8f go = __builtin_amdgcn_wmma_f32_16x16x32_f16(false, a, false, Bg[3], (short)0, z, false, false);

#pragma unroll
        for (int r = 0; r < 8; ++r) {
            const float iv = fast_sigmoid(gi[r]);
            const float fv = fast_sigmoid(gf[r]);
            const float gv = fast_tanh   (gg[r]);
            const float ov = fast_sigmoid(go[r]);
            const float cv = fv * cst[r] + iv * gv;
            cst[r]  = cv;
            hcur[r] = ov * fast_tanh(cv);
        }

        // Publish new h as f16 (rows m = r+8*hi, col n)
#pragma unroll
        for (int r = 0; r < 8; ++r)
            hw[(r + 8 * hi) * 16 + n] = (_Float16)hcur[r];
        __syncthreads();   // single-wave WG: S_NOP + required s_wait_dscnt
    };

    for (int t = 0; t < LSTM_T; ++t) {
        // 16 cells, explicit ping-pong (even count -> parity invariant per t)
#pragma unroll 1
        for (int jj = 0; jj < 8; ++jj) {
            cell(h16A, h16B, 2 * jj);
            cell(h16B, h16A, 2 * jj + 1);
        }

        // pred[m] = sum_n h[m][n]*W_lin[n] + b_lin : 16-lane butterfly
        float pr[8];
#pragma unroll
        for (int r = 0; r < 8; ++r) pr[r] = hcur[r] * wlin;
#pragma unroll
        for (int mask = 1; mask < 16; mask <<= 1) {
#pragma unroll
            for (int r = 0; r < 8; ++r) pr[r] += __shfl_xor(pr[r], mask, 32);
        }
        if (n == 0) {
#pragma unroll
            for (int r = 0; r < 8; ++r) {
                const int m = r + 8 * hi;
                const float pv = pr[r] + blin;
                out[(size_t)(bm + m) * OUTW + LSTM_A + t] = pv;
                win_s[m * WIN_P + start] = pv;          // replace oldest element
            }
        }
        start = (start + 1) & 15;
        __syncthreads();
    }
}

extern "C" void kernel_launch(void* const* d_in, const int* in_sizes, int n_in,
                              void* d_out, int out_size, void* d_ws, size_t ws_size,
                              hipStream_t stream) {
    const float* y     = (const float*)d_in[0];
    const float* u     = (const float*)d_in[1];
    const float* W_ih  = (const float*)d_in[2];
    const float* W_hh  = (const float*)d_in[3];
    const float* b_ih  = (const float*)d_in[4];
    const float* b_hh  = (const float*)d_in[5];
    const float* W_lin = (const float*)d_in[6];
    const float* b_lin = (const float*)d_in[7];
    const float* W_h0  = (const float*)d_in[8];
    const float* b_h0  = (const float*)d_in[9];
    const float* W_c0  = (const float*)d_in[10];
    const float* b_c0  = (const float*)d_in[11];
    float* out = (float*)d_out;

    // 64 batch tiles of 16 rows; one wave32 per block.
    lstm_decoder_wmma_kernel<<<dim3(LSTM_B / 16), dim3(32), 0, stream>>>(
        y, u, W_ih, W_hh, b_ih, b_hh, W_lin, b_lin, W_h0, b_h0, W_c0, b_c0, out);
}
